// LinearAttention_14328010900029
// MI455X (gfx1250) — matrix operand order repaired and checked by reading, not compiled, not run on hardware
//
#include <hip/hip_runtime.h>

#define N_SEQ 2048
#define NSPLIT 4
#define JSPAN (N_SEQ / NSPLIT)     // 512
#define NCHUNK (JSPAN / 32)        // 16
#define LOG2E 1.44269504088896340736f

typedef __attribute__((ext_vector_type(16))) __bf16 v16bf;
typedef __attribute__((ext_vector_type(8)))  float  v8f;
typedef int v4i __attribute__((vector_size(16)));

#if __has_builtin(__builtin_amdgcn_global_load_async_to_lds_b128)
#define HAVE_ASYNC 1
#endif

#ifdef HAVE_ASYNC
// cpol=1 -> TH=NT: one-shot bias stream must not evict the reused working set
#define ASYNC_CPY16(g, l)                                                     \
  __builtin_amdgcn_global_load_async_to_lds_b128(                             \
      (__attribute__((address_space(1))) v4i*)(uintptr_t)(g),                 \
      (__attribute__((address_space(3))) v4i*)(l), 0, 1)
#if __has_builtin(__builtin_amdgcn_s_wait_asynccnt)
#define WAIT_ASYNC(n) __builtin_amdgcn_s_wait_asynccnt(n)
#else
#define WAIT_ASYNC(n) asm volatile("s_wait_asynccnt %0" ::"i"(n) : "memory")
#endif
#else
#define ASYNC_CPY16(g, l)                                                     \
  do { *(uint4*)(l) = *(const uint4*)(g); } while (0)
#define WAIT_ASYNC(n) do {} while (0)
#endif

static __device__ __forceinline__ unsigned short f2bf(float f) {
  union { float f; unsigned int u; } c; c.f = f;
  unsigned int u = c.u;
  return (unsigned short)((u + 0x7FFFu + ((u >> 16) & 1u)) >> 16);
}

// 32 contiguous bf16 -> B fragment (per-lane K run is contiguous)
static __device__ __forceinline__ v16bf load_frag16(const unsigned short* p) {
  union { v16bf v; uint4 q[2]; } u;
  u.q[0] = *reinterpret_cast<const uint4*>(p);
  u.q[1] = *reinterpret_cast<const uint4*>(p + 8);
  return u.v;
}

// A fragment (contiguous K order, matches B): lane (g,m) holds K = g*16+0..15
static __device__ __forceinline__ v16bf load_a_frag(const unsigned short* row, int g) {
  union { v16bf v; uint4 q[2]; } u;
  u.q[0] = *reinterpret_cast<const uint4*>(row + g * 16);
  u.q[1] = *reinterpret_cast<const uint4*>(row + g * 16 + 8);
  return u.v;
}

static __device__ __forceinline__ v8f wmma_bf16(v16bf a, v16bf b, v8f c) {
  return __builtin_amdgcn_wmma_f32_16x16x32_bf16(false, a, false, b, (short)0, c, false, false);
}

// ---------------- Phase 1: per-row feature map + projections (bf16 outputs) ---------------
__global__ __launch_bounds__(64) void prep_rows(
    const float* __restrict__ h, const float* __restrict__ R,
    const float* __restrict__ a_q, const float* __restrict__ a_k,
    const float* __restrict__ W,
    unsigned short* __restrict__ Qb, unsigned short* __restrict__ Kb,
    unsigned short* __restrict__ WhT) {
  const int row = blockIdx.x;
  const int b = row / N_SEQ;
  const int n = row % N_SEQ;
  const int t = threadIdx.x;
  __shared__ float hrow[128];
  __shared__ float phi[128];
  hrow[t]      = h[(size_t)row * 128 + t];
  hrow[t + 64] = h[(size_t)row * 128 + t + 64];
  __syncthreads();
  float pr = 0.f;
  #pragma unroll 8
  for (int f = 0; f < 128; ++f) pr += hrow[f] * R[f * 64 + t];
  phi[t]      = __cosf(pr);
  phi[t + 64] = __sinf(pr);
  __syncthreads();
  float q = 0.f, k = 0.f;
  #pragma unroll 8
  for (int f = 0; f < 128; ++f) {
    float ph = phi[f];
    q += ph * a_q[f * 64 + t];
    k += ph * a_k[f * 64 + t];
  }
  Qb[(size_t)row * 64 + t] = f2bf(q);
  Kb[(size_t)row * 64 + t] = f2bf(k);
  #pragma unroll
  for (int c4 = 0; c4 < 4; ++c4) {
    int c = t + c4 * 64;
    float w = 0.f;
    #pragma unroll 8
    for (int f = 0; f < 128; ++f) w += hrow[f] * W[f * 256 + c];
    WhT[((size_t)b * 256 + c) * N_SEQ + n] = f2bf(w);   // transposed: contiguous in j
  }
}

// output_proj (256,64) -> bf16 transposed (64,256)
__global__ __launch_bounds__(256) void prep_op(const float* __restrict__ OP,
                                               unsigned short* __restrict__ OPbT) {
  int i = blockIdx.x * 256 + threadIdx.x;
  int k = i & 255, c = i >> 8;
  OPbT[(size_t)c * 256 + k] = f2bf(OP[(size_t)k * 64 + c]);
}

// Issue 16 async 16B copies: proj(2KB)+sim(2KB)+quad(4KB) for a 16x32 block -> LDS buffer
static __device__ __forceinline__ void prefetch_bias(
    const float* __restrict__ proj, const float* __restrict__ sim,
    const int2* __restrict__ quad, size_t rowbase, int j0,
    unsigned char* buf, int lane) {
  {
    const int row = lane >> 3, seg = lane & 7;
    #pragma unroll
    for (int i = 0; i < 4; ++i) {
      const int rr = i * 4 + row;
      const char* gp = (const char*)(proj + rowbase + (size_t)rr * N_SEQ + j0) + seg * 16;
      const char* gs = (const char*)(sim + rowbase + (size_t)rr * N_SEQ + j0) + seg * 16;
      ASYNC_CPY16(gp, buf + rr * 128 + seg * 16);
      ASYNC_CPY16(gs, buf + 2048 + rr * 128 + seg * 16);
    }
  }
  {
    const int row = lane >> 4, seg = lane & 15;
    #pragma unroll
    for (int i = 0; i < 8; ++i) {
      const int rr = i * 2 + row;
      const char* gq = (const char*)(quad + rowbase + (size_t)rr * N_SEQ + j0) + seg * 16;
      ASYNC_CPY16(gq, buf + 4096 + rr * 256 + seg * 16);
    }
  }
}

// -------- Phase 2: fused flash-attention over a j-slice; emits unnormalized partials ------
__global__ __launch_bounds__(128) void attn_kernel(
    const float* __restrict__ proj, const float* __restrict__ sim,
    const int2* __restrict__ quad, const float* __restrict__ qbias,
    const unsigned short* __restrict__ Qb, const unsigned short* __restrict__ Kb,
    const unsigned short* __restrict__ WhT,
    float* __restrict__ wsAcc, float* __restrict__ wsStat) {
  // per-wave region: biasbuf0 (8KB) | biasbuf1 (8KB) | P-stage (1KB)
  __shared__ __align__(16) unsigned char ldsraw[4][17408];
  const int split = blockIdx.x & 3;
  const int bt = blockIdx.x >> 2;
  const int b = bt >> 5;
  const int tile = bt & 31;
  const int wave = threadIdx.x >> 5;
  const int lane = threadIdx.x & 31;
  const int g = lane >> 4, m = lane & 15;
  const int r0 = tile * 64 + wave * 16;
  const int jbeg = split * JSPAN;
  unsigned char* buf0 = ldsraw[wave];
  unsigned char* buf1 = ldsraw[wave] + 8192;
  unsigned short* pstage = (unsigned short*)(ldsraw[wave] + 16384);

  // register-resident quadrant bias: lane i holds entry i (i<16); gathered via ds_bpermute
  const float qbv = qbias[lane & 15];

  // Q A-fragments (K 0..31 and 32..63)
  const unsigned short* Qrow = Qb + ((size_t)b * N_SEQ + r0 + m) * 64;
  const v16bf aq0 = load_a_frag(Qrow, g);
  const v16bf aq1 = load_a_frag(Qrow + 32, g);

  const v8f vzero = {0.f, 0.f, 0.f, 0.f, 0.f, 0.f, 0.f, 0.f};
  v8f acc[16];
  #pragma unroll
  for (int o = 0; o < 16; ++o) acc[o] = vzero;
  float mrow[8], lrow[8];
  #pragma unroll
  for (int r = 0; r < 8; ++r) { mrow[r] = -3.0e38f; lrow[r] = 0.f; }

  const size_t rowbase = ((size_t)b * N_SEQ + r0) * (size_t)N_SEQ;

  prefetch_bias(proj, sim, quad, rowbase, jbeg, buf0, lane);

  for (int jc = 0; jc < NCHUNK; ++jc) {
    const int j0 = jbeg + jc * 32;
    unsigned char* cur = (jc & 1) ? buf1 : buf0;
    unsigned char* nxt = (jc & 1) ? buf0 : buf1;

    // K-fragment loads (issue early; L2/WGP$-resident)
    const unsigned short* K0 = Kb + ((size_t)b * N_SEQ + j0 + m) * 64;
    v16bf bk00 = load_frag16(K0 + g * 16);
    v16bf bk01 = load_frag16(K0 + 32 + g * 16);
    v16bf bk10 = load_frag16(K0 + 1024 + g * 16);
    v16bf bk11 = load_frag16(K0 + 1024 + 32 + g * 16);

    if (jc + 1 < NCHUNK) {
      prefetch_bias(proj, sim, quad, rowbase, j0 + 32, nxt, lane);
      WAIT_ASYNC(16);     // the 16 copies for `cur` have completed (in-order)
    } else {
      WAIT_ASYNC(0);
    }

    // S = Q K^T
    v8f S0 = vzero, S1 = vzero;
    S0 = wmma_bf16(aq0, bk00, S0);
    S0 = wmma_bf16(aq1, bk01, S0);
    S1 = wmma_bf16(aq0, bk10, S1);
    S1 = wmma_bf16(aq1, bk11, S1);

    // biases from LDS: batched reads folded straight into sc
    const float* pbuf = (const float*)cur;
    const float* sbuf = (const float*)(cur + 2048);
    const int2* qbuf = (const int2*)(cur + 4096);
    float sc0[8], sc1[8];
    int qi0[8], qi1[8];
    #pragma unroll
    for (int r = 0; r < 8; ++r) {
      const int row = r + 8 * g;
      sc0[r] = S0[r] + pbuf[row * 32 + m] + sbuf[row * 32 + m];
      sc1[r] = S1[r] + pbuf[row * 32 + 16 + m] + sbuf[row * 32 + 16 + m];
      int2 q0 = qbuf[row * 32 + m];
      int2 q1 = qbuf[row * 32 + 16 + m];
      qi0[r] = (q0.x << 2) | q0.y;
      qi1[r] = (q1.x << 2) | q1.y;
    }
    // register-table gather (ds_bpermute), batchable: no LDS dependence
    #pragma unroll
    for (int r = 0; r < 8; ++r) {
      sc0[r] += __shfl(qbv, qi0[r], 32);
      sc1[r] += __shfl(qbv, qi1[r], 32);
    }

    // online softmax over this 32-column chunk
    float p0[8], p1[8];
    v8f av;
    float anyresc = 0.f;
    #pragma unroll
    for (int r = 0; r < 8; ++r) {
      float mx = fmaxf(sc0[r], sc1[r]);
      mx = fmaxf(mx, __shfl_xor(mx, 1, 32));
      mx = fmaxf(mx, __shfl_xor(mx, 2, 32));
      mx = fmaxf(mx, __shfl_xor(mx, 4, 32));
      mx = fmaxf(mx, __shfl_xor(mx, 8, 32));
      float mn = fmaxf(mrow[r], mx);
      float a = exp2f((mrow[r] - mn) * LOG2E);
      mrow[r] = mn;
      p0[r] = exp2f((sc0[r] - mn) * LOG2E);
      p1[r] = exp2f((sc1[r] - mn) * LOG2E);
      float s = p0[r] + p1[r];
      s += __shfl_xor(s, 1, 32);
      s += __shfl_xor(s, 2, 32);
      s += __shfl_xor(s, 4, 32);
      s += __shfl_xor(s, 8, 32);
      lrow[r] = lrow[r] * a + s;
      av[r] = a;
      anyresc += 1.f - a;       // a==1 exactly when row max unchanged
    }
    if (__any(anyresc != 0.f)) {     // wave-uniform: skip identity rescale
      #pragma unroll
      for (int o = 0; o < 16; ++o) acc[o] *= av;
    }

    // C-layout -> A-layout transpose of P through LDS (bf16)
    #pragma unroll
    for (int r = 0; r < 8; ++r) {
      pstage[(r + 8 * g) * 32 + m] = f2bf(p0[r]);
      pstage[(r + 8 * g) * 32 + 16 + m] = f2bf(p1[r]);
    }
    asm volatile("s_wait_dscnt 0x0" ::: "memory");
    v16bf pa = load_a_frag(pstage + m * 32, g);

    // acc += P * Wh
    const unsigned short* wb = WhT + ((size_t)(b * 256 + m)) * N_SEQ + j0 + g * 16;
    #pragma unroll
    for (int o = 0; o < 16; ++o) {
      acc[o] = wmma_bf16(pa, load_frag16(wb), acc[o]);
      wb += (size_t)16 * N_SEQ;
    }
  }

  // emit partials (fragment-linear layout: fully coalesced 32B/lane stores)
  const int rtile = tile * 4 + wave;
  const size_t pidx = (size_t)(b * 128 + rtile) * NSPLIT + split;
  float* pacc = wsAcc + pidx * 4096;
  #pragma unroll
  for (int o = 0; o < 16; ++o)
    *(v8f*)(pacc + o * 256 + lane * 8) = acc[o];
  #pragma unroll
  for (int r = 0; r < 8; ++r)
    if (m == r) {
      wsStat[pidx * 32 + r + 8 * g] = mrow[r];
      wsStat[pidx * 32 + 16 + r + 8 * g] = lrow[r];
    }
}

// -------- Phase 3: merge j-split partials, project 256->64 via WMMA, ELU ------------------
__global__ __launch_bounds__(128) void combine_kernel(
    const float* __restrict__ wsAcc, const float* __restrict__ wsStat,
    const unsigned short* __restrict__ OPbT, float* __restrict__ out) {
  __shared__ __align__(16) unsigned short lds[4][4096];
  const int b = blockIdx.x >> 5;
  const int tile = blockIdx.x & 31;
  const int wave = threadIdx.x >> 5;
  const int lane = threadIdx.x & 31;
  const int g = lane >> 4, m = lane & 15;
  unsigned short* myLds = lds[wave];
  const int rtile = tile * 4 + wave;
  const size_t p0 = (size_t)(b * 128 + rtile) * NSPLIT;

  float ms[NSPLIT][8], ls[NSPLIT][8];
  #pragma unroll
  for (int s = 0; s < NSPLIT; ++s)
    #pragma unroll
    for (int r = 0; r < 8; ++r) {
      ms[s][r] = wsStat[(p0 + s) * 32 + r + 8 * g];
      ls[s][r] = wsStat[(p0 + s) * 32 + 16 + r + 8 * g];
    }
  v8f wv[NSPLIT];
  #pragma unroll
  for (int r = 0; r < 8; ++r) {
    float M = fmaxf(fmaxf(ms[0][r], ms[1][r]), fmaxf(ms[2][r], ms[3][r]));
    float L = 0.f;
    float w[NSPLIT];
    #pragma unroll
    for (int s = 0; s < NSPLIT; ++s) {
      w[s] = exp2f((ms[s][r] - M) * LOG2E);
      L += ls[s][r] * w[s];
    }
    float inv = 1.0f / L;
    #pragma unroll
    for (int s = 0; s < NSPLIT; ++s) wv[s][r] = w[s] * inv;
  }

  #pragma unroll
  for (int o = 0; o < 16; ++o) {
    v8f a = {0.f, 0.f, 0.f, 0.f, 0.f, 0.f, 0.f, 0.f};
    #pragma unroll
    for (int s = 0; s < NSPLIT; ++s) {
      v8f f = *(const v8f*)(wsAcc + (p0 + s) * 4096 + o * 256 + lane * 8);
      a += f * wv[s];
    }
    #pragma unroll
    for (int r = 0; r < 8; ++r)
      myLds[(r + 8 * g) * 256 + o * 16 + m] = f2bf(a[r]);
  }
  asm volatile("s_wait_dscnt 0x0" ::: "memory");

  const v8f vzero = {0.f, 0.f, 0.f, 0.f, 0.f, 0.f, 0.f, 0.f};
  v8f oacc[4];
  #pragma unroll
  for (int f = 0; f < 4; ++f) oacc[f] = vzero;
  #pragma unroll
  for (int kc = 0; kc < 8; ++kc) {
    v16bf pa2 = load_a_frag(myLds + m * 256 + kc * 32, g);
    #pragma unroll
    for (int f = 0; f < 4; ++f) {
      v16bf bop = load_frag16(OPbT + (size_t)(f * 16 + m) * 256 + kc * 32 + g * 16);
      oacc[f] = wmma_bf16(pa2, bop, oacc[f]);
    }
  }

  float* orow = out + ((size_t)b * N_SEQ + tile * 64 + wave * 16 + 8 * g) * 64 + m;
  #pragma unroll
  for (int f = 0; f < 4; ++f)
    #pragma unroll
    for (int r = 0; r < 8; ++r) {
      float x = oacc[f][r];
      orow[(size_t)r * 64 + f * 16] = (x > 0.f) ? x : (exp2f(x * LOG2E) - 1.0f);
    }
}

extern "C" void kernel_launch(void* const* d_in, const int* in_sizes, int n_in,
                              void* d_out, int out_size, void* d_ws, size_t ws_size,
                              hipStream_t stream) {
  (void)in_sizes; (void)n_in; (void)out_size; (void)ws_size;
  const int Bc = 4;
  const float* h    = (const float*)d_in[0];
  const float* proj = (const float*)d_in[1];
  const float* sim  = (const float*)d_in[2];
  const float* W    = (const float*)d_in[3];
  const float* R    = (const float*)d_in[4];
  const float* a_q  = (const float*)d_in[5];
  const float* a_k  = (const float*)d_in[6];
  const float* qb   = (const float*)d_in[7];
  const float* OP   = (const float*)d_in[8];
  const int2*  quad = (const int2*)d_in[9];

  char* ws = (char*)d_ws;
  unsigned short* Qb   = (unsigned short*)(ws);                           // 1 MB
  unsigned short* Kb   = (unsigned short*)(ws + (1u << 20));              // 1 MB
  unsigned short* WhT  = (unsigned short*)(ws + (2u << 20));              // 4 MB
  unsigned short* OPbT = (unsigned short*)(ws + (6u << 20));              // 32 KB
  float* wsStat = (float*)(ws + (6u << 20) + 65536);                      // 256 KB
  float* wsAcc  = (float*)(ws + (8u << 20));                              // 32 MB

  prep_rows<<<Bc * N_SEQ, 64, 0, stream>>>(h, R, a_q, a_k, W, Qb, Kb, WhT);
  prep_op<<<64, 256, 0, stream>>>(OP, OPbT);
  attn_kernel<<<Bc * 32 * NSPLIT, 128, 0, stream>>>(proj, sim, quad, qb, Qb, Kb, WhT,
                                                    wsAcc, wsStat);
  combine_kernel<<<Bc * 32, 128, 0, stream>>>(wsAcc, wsStat, OPbT, (float*)d_out);
}
